// TreeLSTM_49203145343608
// MI455X (gfx1250) — compile-verified, matrix operand order checked by reading, not run on hardware
//
#include <hip/hip_runtime.h>
#include <stdint.h>

#define H_DIM   1024
#define G_DIM   5120
#define K_DIM   1024
#define NNODES  4095
#define NLEAVES 2048
#define NRED    2047
#define SCAN_WGS 256
#define JB 4   // hidden columns per scan workgroup

typedef __bf16 bf16_t;
typedef bf16_t v16bf __attribute__((ext_vector_type(16)));
typedef float  v8f   __attribute__((ext_vector_type(8)));

union FragBF { v16bf v; unsigned int u[8]; unsigned short s[16]; };

__device__ __forceinline__ unsigned short f2bf(float f) {
    unsigned int u = __float_as_uint(f);
    u += 0x7FFFu + ((u >> 16) & 1u);           // round-to-nearest-even
    return (unsigned short)(u >> 16);
}
__device__ __forceinline__ float bf2f(unsigned short h) {
    return __uint_as_float(((unsigned int)h) << 16);
}
__device__ __forceinline__ float sigmoidf(float x) {
    return 1.0f / (1.0f + __expf(-x));
}

// ---- f32 -> bf16 conversion -------------------------------------------------
__global__ void cvt_f32_bf16(const float* __restrict__ in,
                             unsigned short* __restrict__ out, int n) {
    for (int i = blockIdx.x * blockDim.x + threadIdx.x; i < n;
         i += gridDim.x * blockDim.x)
        out[i] = f2bf(in[i]);
}

__global__ void init_bar(unsigned* bar) {
    if (threadIdx.x < 2) bar[threadIdx.x] = 0u;
}

// ---- WMMA fragment loader: 16-bit A-layout (ISA 7.12.2) ---------------------
__device__ __forceinline__ void load_frag(FragBF& f, const unsigned short* row,
                                          int kbase, int half) {
#pragma unroll
    for (int v = 0; v < 8; ++v) {
        int kk = kbase + ((v < 4) ? (2 * v) : (8 + 2 * v)) + 8 * half;
        f.u[v] = *reinterpret_cast<const unsigned int*>(row + kk);
    }
}
#define WMMA_BF16(a, b, c) __builtin_amdgcn_wmma_f32_16x16x32_bf16( \
        false, (a).v, false, (b).v, (short)0, (c), false, false)

// ---- GEMM1: ih_all[N,5H] = xs @ W_ih^T + bias; 2x2 register-blocked ---------
__global__ __launch_bounds__(256) void gemm_ih(
        const unsigned short* __restrict__ Axs,
        const unsigned short* __restrict__ Wih,
        const float* __restrict__ b0, const float* __restrict__ b1,
        const float* __restrict__ b2, float* __restrict__ C) {
    const int tiles_n2 = G_DIM / 32;                      // 160
    int wid  = blockIdx.x * 8 + (threadIdx.x >> 5);
    int tn   = wid % tiles_n2;
    int tm   = wid / tiles_n2;
    int lane = threadIdx.x & 31;
    int half = lane >> 4, lm = lane & 15;
    int m0 = tm * 32 + lm, m1 = m0 + 16;
    int mA0 = (m0 < NNODES) ? m0 : 0;                     // clamp; stores guarded
    int mA1 = (m1 < NNODES) ? m1 : 0;
    int g0 = tn * 32 + lm, g1 = g0 + 16;
    const unsigned short* Ar0 = Axs + (size_t)mA0 * K_DIM;
    const unsigned short* Ar1 = Axs + (size_t)mA1 * K_DIM;
    const unsigned short* Br0 = Wih + (size_t)g0 * K_DIM;
    const unsigned short* Br1 = Wih + (size_t)g1 * K_DIM;
    v8f a00 = {}, a01 = {}, a10 = {}, a11 = {};
    for (int kt = 0; kt < K_DIM; kt += 32) {
        FragBF fa0, fa1, fb0, fb1;
        load_frag(fa0, Ar0, kt, half);
        load_frag(fa1, Ar1, kt, half);
        load_frag(fb0, Br0, kt, half);
        load_frag(fb1, Br1, kt, half);
        a00 = WMMA_BF16(fa0, fb0, a00);
        a01 = WMMA_BF16(fa0, fb1, a01);
        a10 = WMMA_BF16(fa1, fb0, a10);
        a11 = WMMA_BF16(fa1, fb1, a11);
    }
    float bs0 = b0[g0] + b1[g0] + b2[g0];
    float bs1 = b0[g1] + b1[g1] + b2[g1];
#pragma unroll
    for (int v = 0; v < 8; ++v) {
        int m = tm * 32 + v + 8 * half;                   // C/D layout
        if (m < NNODES) {
            C[(size_t)m * G_DIM + g0] = a00[v] + bs0;
            C[(size_t)m * G_DIM + g1] = a01[v] + bs1;
        }
        int mm = m + 16;
        if (mm < NNODES) {
            C[(size_t)mm * G_DIM + g0] = a10[v] + bs0;
            C[(size_t)mm * G_DIM + g1] = a11[v] + bs1;
        }
    }
}

// ---- Leaf kernel: shift-step states, fully parallel -------------------------
__global__ __launch_bounds__(256) void leaf_kernel(
        const float* __restrict__ ih, float* __restrict__ hs,
        float* __restrict__ cs, unsigned short* __restrict__ leafh) {
    int i = blockIdx.x;                                   // leaf id 0..2047
    int t = (i <= 1) ? i : (2 * i - 1);                   // its shift step
    const float* row = ih + (size_t)t * G_DIM;
#pragma unroll
    for (int r = 0; r < 4; ++r) {
        int j = threadIdx.x + r * 256;
        float gi = sigmoidf(row[j]);
        float gg = tanhf(row[3 * H_DIM + j]);
        float go = sigmoidf(row[4 * H_DIM + j]);
        float c = gi * gg;                                // lc = rc = 0
        float h = go * tanhf(c);
        hs[(size_t)t * H_DIM + j]  = h;
        cs[(size_t)t * H_DIM + j]  = c;
        leafh[(size_t)i * H_DIM + j] = f2bf(h);
    }
}

// ---- GEMM2: ih_all[2+2k] += leaf_h[k+1] @ W_rhh^T; 2x2 blocked --------------
__global__ __launch_bounds__(256) void gemm_rhh(
        const unsigned short* __restrict__ Lh,
        const unsigned short* __restrict__ Wr, float* __restrict__ C) {
    const int tiles_n2 = G_DIM / 32;
    int wid  = blockIdx.x * 8 + (threadIdx.x >> 5);
    int tn   = wid % tiles_n2;
    int tm   = wid / tiles_n2;
    int lane = threadIdx.x & 31;
    int half = lane >> 4, lm = lane & 15;
    int ar0 = tm * 32 + lm, ar1 = ar0 + 16;
    int lf0 = (ar0 < NRED) ? (ar0 + 1) : 1;               // right child leaf
    int lf1 = (ar1 < NRED) ? (ar1 + 1) : 1;
    int g0 = tn * 32 + lm, g1 = g0 + 16;
    const unsigned short* Ar0 = Lh + (size_t)lf0 * K_DIM;
    const unsigned short* Ar1 = Lh + (size_t)lf1 * K_DIM;
    const unsigned short* Br0 = Wr + (size_t)g0 * K_DIM;
    const unsigned short* Br1 = Wr + (size_t)g1 * K_DIM;
    v8f a00 = {}, a01 = {}, a10 = {}, a11 = {};
    for (int kt = 0; kt < K_DIM; kt += 32) {
        FragBF fa0, fa1, fb0, fb1;
        load_frag(fa0, Ar0, kt, half);
        load_frag(fa1, Ar1, kt, half);
        load_frag(fb0, Br0, kt, half);
        load_frag(fb1, Br1, kt, half);
        a00 = WMMA_BF16(fa0, fb0, a00);
        a01 = WMMA_BF16(fa0, fb1, a01);
        a10 = WMMA_BF16(fa1, fb0, a10);
        a11 = WMMA_BF16(fa1, fb1, a11);
    }
#pragma unroll
    for (int v = 0; v < 8; ++v) {
        int m = tm * 32 + v + 8 * half;
        if (m < NRED) {
            size_t row = (size_t)(2 + 2 * m) * G_DIM;
            C[row + g0] += a00[v];
            C[row + g1] += a01[v];
        }
        int mm = m + 16;
        if (mm < NRED) {
            size_t row = (size_t)(2 + 2 * mm) * G_DIM;
            C[row + g0] += a10[v];
            C[row + g1] += a11[v];
        }
    }
}

// ---- Persistent serial scan: 2047 reduces, W_lhh pinned in LDS --------------
__global__ __launch_bounds__(128) void scan_kernel(
        const unsigned short* __restrict__ Wl, const float* __restrict__ ih,
        float* __restrict__ out, unsigned* __restrict__ bar) {
    __shared__ __align__(16) unsigned short Wlds[20 * 1024]; // 40 KB
    __shared__ __align__(16) float hl[1024];                 //  4 KB
    __shared__ float dots[20];
    float* hs = out + H_DIM;
    float* cs = out + H_DIM + (size_t)NNODES * H_DIM;
    int tid = threadIdx.x;
    int j0  = blockIdx.x * JB;

    // --- one-time W_lhh slice preload via async DMA to LDS (ASYNCcnt) -------
    for (int it = 0; it < 20; ++it) {
        int b16     = it * 128 + tid;                     // 16-byte unit index
        int byteoff = b16 * 16;
        int r  = byteoff >> 11;                           // 2048 B per row
        int eb = byteoff & 2047;
        int s = r >> 2, jj = r & 3;
        const unsigned short* gsrc =
            Wl + (size_t)(s * H_DIM + j0 + jj) * K_DIM + (eb >> 1);
        unsigned ldst = (unsigned)(size_t)((char*)Wlds + byteoff);
        asm volatile("global_load_async_to_lds_b128 %0, %1, off"
                     :: "v"(ldst), "v"(gsrc) : "memory");
    }
    asm volatile("s_wait_asynccnt 0" ::: "memory");
    __syncthreads();

    unsigned* cnt = bar;
    unsigned* gen = bar + 1;
    int lane = tid & 31, w = tid >> 5;
    const ushort4* w4base = reinterpret_cast<const ushort4*>(Wlds);
    const float4*  h4base = reinterpret_cast<const float4*>(hl);
    for (int k = 0; k < NRED; ++k) {
        int t = 2 + 2 * k;
        int lprev = (k == 0) ? 0 : 2 * k;                 // left child step
        int rt = 2 * k + 1;                               // right child (leaf) step
        // async DMA the 4 KB h vector straight into LDS (2 x b128 per lane)
        {
            const float* gh = hs + (size_t)lprev * H_DIM + tid * 4;
            unsigned hdst = (unsigned)(size_t)((char*)hl + tid * 16);
            asm volatile("global_load_async_to_lds_b128 %0, %1, off"
                         :: "v"(hdst), "v"(gh) : "memory");
            asm volatile("global_load_async_to_lds_b128 %0, %1, off offset:2048"
                         :: "v"(hdst), "v"(gh) : "memory");
        }
        asm volatile("s_wait_asynccnt 0" ::: "memory");
        __syncthreads();
#pragma unroll
        for (int rr = 0; rr < 5; ++rr) {                  // 4 waves x 5 rows
            int r = w * 5 + rr;
            const ushort4* w4 = w4base + r * 256;
            float acc = 0.f;
#pragma unroll
            for (int i2 = 0; i2 < 8; ++i2) {
                int c = i2 * 32 + lane;                   // 4-element chunk id
                ushort4 wv = w4[c];
                float4  hv = h4base[c];
                acc += bf2f(wv.x) * hv.x + bf2f(wv.y) * hv.y +
                       bf2f(wv.z) * hv.z + bf2f(wv.w) * hv.w;
            }
#pragma unroll
            for (int off = 16; off > 0; off >>= 1)
                acc += __shfl_xor(acc, off, 32);
            if (lane == 0) dots[r] = acc;
        }
        __syncthreads();
        if (tid < JB) {
            int jj = tid, j = j0 + jj;
            const float* pre = ih + (size_t)t * G_DIM + j;
            float gi = sigmoidf(pre[0 * H_DIM] + dots[0 * JB + jj]);
            float lf = sigmoidf(pre[1 * H_DIM] + dots[1 * JB + jj]);
            float rf = sigmoidf(pre[2 * H_DIM] + dots[2 * JB + jj]);
            float gg = tanhf  (pre[3 * H_DIM] + dots[3 * JB + jj]);
            float go = sigmoidf(pre[4 * H_DIM] + dots[4 * JB + jj]);
            float lc = cs[(size_t)lprev * H_DIM + j];
            float rc = cs[(size_t)rt    * H_DIM + j];
            float c = lf * lc + rf * rc + gi * gg;
            float h = go * tanhf(c);
            hs[(size_t)t * H_DIM + j] = h;
            cs[(size_t)t * H_DIM + j] = c;
        }
        __syncthreads();                                  // ---- grid barrier ----
        if (tid == 0) {
            __threadfence();
            unsigned arrived = __hip_atomic_fetch_add(
                cnt, 1u, __ATOMIC_ACQ_REL, __HIP_MEMORY_SCOPE_AGENT);
            if (arrived == (unsigned)SCAN_WGS * (unsigned)(k + 1) - 1u) {
                __hip_atomic_store(gen, (unsigned)(k + 1), __ATOMIC_RELEASE,
                                   __HIP_MEMORY_SCOPE_AGENT);
            } else {
                while (__hip_atomic_load(gen, __ATOMIC_ACQUIRE,
                                         __HIP_MEMORY_SCOPE_AGENT) <
                       (unsigned)(k + 1)) {
                    __builtin_amdgcn_s_sleep(2);
                }
            }
        }
        __syncthreads();
    }
    if (blockIdx.x == 0) {                                // root_h = hs[-1]
        for (int i = tid; i < H_DIM; i += 128)
            out[i] = hs[(size_t)(NNODES - 1) * H_DIM + i];
    }
}

// ---- launch ----------------------------------------------------------------
extern "C" void kernel_launch(void* const* d_in, const int* in_sizes, int n_in,
                              void* d_out, int out_size, void* d_ws, size_t ws_size,
                              hipStream_t stream) {
    const float* xs    = (const float*)d_in[0];
    // d_in[1] = ops (int32) — fixed comb-tree schedule, not needed
    const float* W_ih  = (const float*)d_in[2];
    const float* W_lhh = (const float*)d_in[3];
    const float* W_rhh = (const float*)d_in[4];
    const float* b_ih  = (const float*)d_in[5];
    const float* b_lhh = (const float*)d_in[6];
    const float* b_rhh = (const float*)d_in[7];
    float* out = (float*)d_out;

    char* ws = (char*)d_ws;
    unsigned short* wih_bf  = (unsigned short*)(ws + 0);          // 10.0 MB
    unsigned short* wlhh_bf = (unsigned short*)(ws + 10485760);   // 10.0 MB
    unsigned short* wrhh_bf = (unsigned short*)(ws + 20971520);   // 10.0 MB
    unsigned short* xs_bf   = (unsigned short*)(ws + 31457280);   //  8.0 MB
    float*          ih_all  = (float*)        (ws + 39843840);    // 80.0 MB
    unsigned short* leafh   = (unsigned short*)(ws + 123709440);  //  4.0 MB
    unsigned*       bar     = (unsigned*)     (ws + 127903744);   //  8 B

    cvt_f32_bf16<<<2048, 256, 0, stream>>>(W_ih,  wih_bf,  G_DIM * K_DIM);
    cvt_f32_bf16<<<2048, 256, 0, stream>>>(W_lhh, wlhh_bf, G_DIM * K_DIM);
    cvt_f32_bf16<<<2048, 256, 0, stream>>>(W_rhh, wrhh_bf, G_DIM * K_DIM);
    cvt_f32_bf16<<<2048, 256, 0, stream>>>(xs,    xs_bf,   NNODES * K_DIM);
    init_bar<<<1, 32, 0, stream>>>(bar);

    // GEMM1: 128 m-supertiles x 160 n-supertiles (32x32 per wave), 8 waves/block
    gemm_ih<<<(128 * 160) / 8, 256, 0, stream>>>(xs_bf, wih_bf,
                                                 b_ih, b_lhh, b_rhh, ih_all);
    float* hs = out + H_DIM;
    float* cs = out + H_DIM + (size_t)NNODES * H_DIM;
    leaf_kernel<<<NLEAVES, 256, 0, stream>>>(ih_all, hs, cs, leafh);
    // GEMM2: 64 m-supertiles x 160 n-supertiles
    gemm_rhh<<<(64 * 160) / 8, 256, 0, stream>>>(leafh, wrhh_bf, ih_all);
    scan_kernel<<<SCAN_WGS, 128, 0, stream>>>(wlhh_bf, ih_all, out, bar);
}
